// PTGCN_82944408420563
// MI455X (gfx1250) — matrix operand order repaired and checked by reading, not compiled, not run on hardware
//
#include <hip/hip_runtime.h>
#include <math.h>

#define TBLW 50
#define KNB  20
#define DD   128
#define EEH  256
#define KV_STR 264   // halfs, padded row stride for 32x256 KV tile
#define KH_STR 264   // halfs, padded row stride for K/V outputs
#define BLK  256

typedef _Float16 v16h __attribute__((ext_vector_type(16)));
typedef _Float16 v8h  __attribute__((ext_vector_type(8)));
typedef float    v8f  __attribute__((ext_vector_type(8)));

#define WMMA_F16(A, Bm, C) \
  __builtin_amdgcn_wmma_f32_16x16x32_f16(false, (A), false, (Bm), (short)0, (C), false, false)

__device__ __forceinline__ v16h make_frag(v8h lo, v8h hi) {
  v16h r;
#pragma unroll
  for (int i = 0; i < 8; ++i) { r[i] = lo[i]; r[i + 8] = hi[i]; }
  return r;
}

// A fragment (16x32 f16): lane L holds row M = mt*16 + (L%16),
// K chunks [kb..kb+7] and [kb+16..kb+23], kb = ks*32 + (L/16)*8.
__device__ __forceinline__ v16h lds_a_frag(const _Float16* kv, int mt, int ks, int lane) {
  const int m  = mt * 16 + (lane & 15);
  const int kb = ks * 32 + ((lane >> 4) << 3);
  const _Float16* p = kv + m * KV_STR + kb;
  return make_frag(*(const v8h*)p, *(const v8h*)(p + 16));
}

// B fragment (32x16 f16) for D = KV @ W^T: B[k][n] = W[n][k] (row-major W [256x256] f16).
// Lane L holds column n = nt*16 + (L%16), K chunks [kb..kb+7] and [kb+16..kb+23], kb = ks*32 + (L/16)*8.
__device__ __forceinline__ v16h glb_b_frag(const _Float16* W, int nt, int ks, int lane) {
  const int j = nt * 16 + (lane & 15);
  const int e = ks * 32 + ((lane >> 4) << 3);
  const _Float16* p = W + (size_t)j * EEH + e;
  return make_frag(*(const v8h*)p, *(const v8h*)(p + 16));
}

// dot of fp32 LDS vector with fp16 weight row (16B vector loads)
__device__ __forceinline__ float dot_h(const float* x, const _Float16* w, int n) {
  float acc = 0.f;
  for (int e = 0; e < n; e += 8) {
    v8h w8 = *(const v8h*)(w + e);
#pragma unroll
    for (int i = 0; i < 8; ++i) acc += x[e + i] * (float)w8[i];
  }
  return acc;
}

struct AttnW {
  const _Float16 *Wq, *Wk, *Wv, *Wo, *W1, *W2;   // fp16 (converted) weights
  const float *bq, *bk, *bv, *bo, *b1, *b2;      // fp32 biases
  const float *tw, *tb;                          // time encode w/b [128]
};

// One block (256 threads / 8 waves) = one target node's full attention layer.
__device__ void attend_core(const AttnW& w, int l,
                            const float* __restrict__ nf_row,
                            const float* const* __restrict__ nbr_rows,
                            const float* __restrict__ dts,
                            const int* __restrict__ mks,
                            float* __restrict__ out_row)
{
  __shared__ __align__(16) _Float16 kvA[32 * KV_STR];   // KV tile (M padded to 32)
  __shared__ __align__(16) _Float16 khs[KNB * KH_STR];  // key projections
  __shared__ __align__(16) _Float16 vhs[KNB * KH_STR];  // value projections
  __shared__ float qf[EEH];     // query input  (nf || tenc(0))
  __shared__ float qh[EEH];     // projected query
  __shared__ float nfv[DD];
  __shared__ float scs[2 * KNB];
  __shared__ float aw[2 * KNB];
  __shared__ float ovec[EEH];
  __shared__ float o2[EEH];
  __shared__ float h1[DD];
  __shared__ int   anyv;

  const int tid  = threadIdx.x;
  const int lane = tid & 31;
  const int wv   = tid >> 5;

  // ---- build query input and KV tile in LDS ----
  if (tid < DD) {
    const float v = nf_row[tid];
    nfv[tid] = v;
    qf[tid]  = v;
    qf[DD + tid] = cosf(w.tb[tid]);           // tenc(0) = cos(0*w + b)
  }
  if (tid == 0) {
    int a = 0;
#pragma unroll
    for (int k = 0; k < KNB; ++k) a |= mks[k];
    anyv = a;
  }
  for (int idx = tid; idx < KNB * EEH; idx += BLK) {
    const int k = idx >> 8, d = idx & 255;
    float v;
    if (d < DD) {
      v = nbr_rows[k][d];
    } else {
      const int dd = d - DD;
      v = cosf(dts[k] * w.tw[dd] + w.tb[dd]);
    }
    kvA[k * KV_STR + d] = (_Float16)v;
  }
  for (int idx = tid; idx < 12 * EEH; idx += BLK) {   // zero pad rows 20..31
    const int k = KNB + (idx >> 8), d = idx & 255;
    kvA[k * KV_STR + d] = (_Float16)0.f;
  }
  __syncthreads();

  // ---- query projection (small matvec, VALU): qh[j] = bq + qf . Wq[j,:] ----
  {
    const int j = tid;
    qh[j] = w.bq[l * EEH + j] + dot_h(qf, w.Wq + (size_t)l * EEH * EEH + (size_t)j * EEH, EEH);
  }

  // ---- K/V projections via WMMA: [32x256] @ [256x256]^T, twice ----
  {
    const _Float16* Wk = w.Wk + (size_t)l * EEH * EEH;
    const _Float16* Wv = w.Wv + (size_t)l * EEH * EEH;
    const v8f z = {0.f, 0.f, 0.f, 0.f, 0.f, 0.f, 0.f, 0.f};
    v8f aK[2][2] = {{z, z}, {z, z}};
    v8f aV[2][2] = {{z, z}, {z, z}};
    const int nt0 = wv * 2;                   // each wave: 2 N-tiles x 2 M-tiles
#pragma unroll
    for (int ks = 0; ks < 8; ++ks) {
      const v16h a0 = lds_a_frag(kvA, 0, ks, lane);
      const v16h a1 = lds_a_frag(kvA, 1, ks, lane);
#pragma unroll
      for (int j = 0; j < 2; ++j) {
        const v16h bK = glb_b_frag(Wk, nt0 + j, ks, lane);
        const v16h bV = glb_b_frag(Wv, nt0 + j, ks, lane);
        aK[0][j] = WMMA_F16(a0, bK, aK[0][j]);
        aK[1][j] = WMMA_F16(a1, bK, aK[1][j]);
        aV[0][j] = WMMA_F16(a0, bV, aV[0][j]);
        aV[1][j] = WMMA_F16(a1, bV, aV[1][j]);
      }
    }
    // D layout: lane L, vgpr r -> m = mt*16 + (L/16)*8 + r, n = nt*16 + (L%16)
#pragma unroll
    for (int mt = 0; mt < 2; ++mt) {
      const int mb = mt * 16 + ((lane >> 4) << 3);
#pragma unroll
      for (int j = 0; j < 2; ++j) {
        const int n = (nt0 + j) * 16 + (lane & 15);
        const float bkv = w.bk[l * EEH + n];
        const float bvv = w.bv[l * EEH + n];
#pragma unroll
        for (int r = 0; r < 8; ++r) {
          const int m = mb + r;
          if (m < KNB) {
            khs[m * KH_STR + n] = (_Float16)(aK[mt][j][r] + bkv);
            vhs[m * KH_STR + n] = (_Float16)(aV[mt][j][r] + bvv);
          }
        }
      }
    }
  }
  __syncthreads();

  // ---- attention scores: sc[h,k] = qh[h,:] . kh[k,h,:] / sqrt(128), masked ----
  if (tid < 2 * KNB) {
    const int h = tid / KNB, k = tid % KNB;
    const _Float16* kr = khs + k * KH_STR + h * DD;
    const float*    qr = qh + h * DD;
    float acc = 0.f;
#pragma unroll 8
    for (int d = 0; d < DD; ++d) acc += qr[d] * (float)kr[d];
    const float s = acc * 0.0883883476483f;   // 1/sqrt(128)
    scs[tid] = mks[k] ? s : -1e9f;
  }
  __syncthreads();
  if (tid < 2) {
    const int h = tid;
    float mx = -3.0e38f;
#pragma unroll
    for (int k = 0; k < KNB; ++k) mx = fmaxf(mx, scs[h * KNB + k]);
    float s = 0.f;
#pragma unroll
    for (int k = 0; k < KNB; ++k) { const float e = expf(scs[h * KNB + k] - mx); aw[h * KNB + k] = e; s += e; }
    const float inv = 1.f / s;
#pragma unroll
    for (int k = 0; k < KNB; ++k) aw[h * KNB + k] *= inv;
  }
  __syncthreads();

  // ---- o[h*128+d] = sum_k a[h,k] * vh[k,h,d] ----
  {
    const int h = tid >> 7;
    float acc = 0.f;
#pragma unroll
    for (int k = 0; k < KNB; ++k) acc += aw[h * KNB + k] * (float)vhs[k * KH_STR + tid];
    ovec[tid] = acc;
  }
  __syncthreads();

  // ---- o @ Wo^T + bo, zero all-invalid rows ----
  {
    const int j = tid;
    const float acc = w.bo[l * EEH + j] + dot_h(ovec, w.Wo + (size_t)l * EEH * EEH + (size_t)j * EEH, EEH);
    o2[j] = anyv ? acc : 0.f;
  }
  __syncthreads();

  // ---- h1 = relu([o2 || nf] @ W1^T + b1) ----
  if (tid < DD) {
    const int j = tid;
    const _Float16* wr = w.W1 + (size_t)l * DD * (EEH + DD) + (size_t)j * (EEH + DD);
    float acc = w.b1[l * DD + j];
    acc += dot_h(o2, wr, EEH);
    acc += dot_h(nfv, wr + EEH, DD);
    h1[j] = fmaxf(acc, 0.f);
  }
  __syncthreads();

  // ---- out = h1 @ W2^T + b2 ----
  if (tid < DD) {
    const int j = tid;
    out_row[j] = w.b2[l * DD + j] +
                 dot_h(h1, w.W2 + (size_t)l * DD * DD + (size_t)j * DD, DD);
  }
}

__global__ void ptgcn_cvt_f16(const float* __restrict__ s, _Float16* __restrict__ d, int n) {
  const int i = blockIdx.x * BLK + threadIdx.x;
  if (i < n) d[i] = (_Float16)s[i];
}

// Inner layer: one block per (batch b, outer-slot ko) item node.
__global__ __launch_bounds__(BLK)
void ptgcn_inner_kernel(AttnW w,
                        const int* __restrict__ edges,
                        const int* __restrict__ user_neig,
                        const int* __restrict__ user_edges,
                        const int* __restrict__ user_times,
                        const int* __restrict__ item_neig,
                        const int* __restrict__ item_times,
                        const int* __restrict__ item_mask,
                        const float* __restrict__ user_emb,
                        const float* __restrict__ item_emb,
                        float* __restrict__ nbrbuf)
{
  __shared__ const float* sp[KNB];
  __shared__ float sd[KNB];
  __shared__ int   sm[KNB];
  __shared__ int   snode;
  const int b2  = blockIdx.x;
  const int tid = threadIdx.x;
  if (tid < KNB) {
    const int b    = b2 / KNB;
    const int ko   = b2 % KNB;
    const int base = edges[b] * TBLW + (TBLW - KNB) + ko;
    const int ee   = user_edges[base];
    const int tt   = user_times[base];
    const int base2 = ee * TBLW + (TBLW - KNB) + tid;
    sp[tid] = user_emb + (size_t)item_neig[base2] * DD;
    sd[tid] = (float)(tt - item_times[base2]);
    sm[tid] = item_mask[base2];
    if (tid == 0) snode = user_neig[base];
  }
  __syncthreads();
  attend_core(w, 0, item_emb + (size_t)snode * DD, sp, sd, sm,
              nbrbuf + (size_t)b2 * DD);
}

// Outer layer: one block per batch user node; neighbor features from nbrbuf.
__global__ __launch_bounds__(BLK)
void ptgcn_outer_kernel(AttnW w,
                        const int* __restrict__ nodes,
                        const int* __restrict__ edges,
                        const int* __restrict__ timestamps,
                        const int* __restrict__ user_times,
                        const int* __restrict__ user_mask,
                        const float* __restrict__ user_emb,
                        const float* __restrict__ nbrbuf,
                        float* __restrict__ out)
{
  __shared__ const float* sp[KNB];
  __shared__ float sd[KNB];
  __shared__ int   sm[KNB];
  __shared__ int   snode;
  const int b   = blockIdx.x;
  const int tid = threadIdx.x;
  if (tid < KNB) {
    const int base = edges[b] * TBLW + (TBLW - KNB) + tid;
    sd[tid] = (float)(timestamps[b] - user_times[base]);
    sm[tid] = user_mask[base];
    sp[tid] = nbrbuf + (size_t)(b * KNB + tid) * DD;
    if (tid == 0) snode = nodes[b];
  }
  __syncthreads();
  attend_core(w, 1, user_emb + (size_t)snode * DD, sp, sd, sm,
              out + (size_t)b * DD);
}

extern "C" void kernel_launch(void* const* d_in, const int* in_sizes, int n_in,
                              void* d_out, int out_size, void* d_ws, size_t ws_size,
                              hipStream_t stream)
{
  const int*   nodes      = (const int*)d_in[0];
  const int*   edges      = (const int*)d_in[1];
  const int*   timestamps = (const int*)d_in[2];
  const int*   user_neig  = (const int*)d_in[4];
  const int*   user_edges = (const int*)d_in[5];
  const int*   user_times = (const int*)d_in[6];
  const int*   user_mask  = (const int*)d_in[7];
  const int*   item_neig  = (const int*)d_in[8];
  const int*   item_times = (const int*)d_in[10];
  const int*   item_mask  = (const int*)d_in[11];
  const float* user_emb   = (const float*)d_in[12];
  const float* item_emb   = (const float*)d_in[13];
  const int B = in_sizes[0];

  // workspace layout (halfs): Wq|Wk|Wv|Wo (2*256*256 each), W1 (2*128*384), W2 (2*128*128), then f32 nbrbuf
  _Float16* wsH = (_Float16*)d_ws;
  _Float16* WqH = wsH;
  _Float16* WkH = wsH + 131072;
  _Float16* WvH = wsH + 262144;
  _Float16* WoH = wsH + 393216;
  _Float16* W1H = wsH + 524288;
  _Float16* W2H = wsH + 622592;
  float* nbrbuf = (float*)(wsH + 655360);

  const int nQ = 2 * EEH * EEH;        // 131072
  const int n1 = 2 * DD * (EEH + DD);  // 98304
  const int n2 = 2 * DD * DD;          // 32768
  ptgcn_cvt_f16<<<(nQ + BLK - 1) / BLK, BLK, 0, stream>>>((const float*)d_in[16], WqH, nQ);
  ptgcn_cvt_f16<<<(nQ + BLK - 1) / BLK, BLK, 0, stream>>>((const float*)d_in[18], WkH, nQ);
  ptgcn_cvt_f16<<<(nQ + BLK - 1) / BLK, BLK, 0, stream>>>((const float*)d_in[20], WvH, nQ);
  ptgcn_cvt_f16<<<(nQ + BLK - 1) / BLK, BLK, 0, stream>>>((const float*)d_in[22], WoH, nQ);
  ptgcn_cvt_f16<<<(n1 + BLK - 1) / BLK, BLK, 0, stream>>>((const float*)d_in[24], W1H, n1);
  ptgcn_cvt_f16<<<(n2 + BLK - 1) / BLK, BLK, 0, stream>>>((const float*)d_in[26], W2H, n2);

  AttnW w;
  w.Wq = WqH; w.Wk = WkH; w.Wv = WvH; w.Wo = WoH; w.W1 = W1H; w.W2 = W2H;
  w.bq = (const float*)d_in[17]; w.bk = (const float*)d_in[19];
  w.bv = (const float*)d_in[21]; w.bo = (const float*)d_in[23];
  w.b1 = (const float*)d_in[25]; w.b2 = (const float*)d_in[27];
  w.tw = (const float*)d_in[14]; w.tb = (const float*)d_in[15];

  ptgcn_inner_kernel<<<B * KNB, BLK, 0, stream>>>(w, edges, user_neig, user_edges, user_times,
                                                  item_neig, item_times, item_mask,
                                                  user_emb, item_emb, nbrbuf);
  ptgcn_outer_kernel<<<B, BLK, 0, stream>>>(w, nodes, edges, timestamps, user_times, user_mask,
                                            user_emb, nbrbuf, (float*)d_out);
}
